// Forward_78211354460854
// MI455X (gfx1250) — matrix operand order repaired and checked by reading, not compiled, not run on hardware
//
#include <hip/hip_runtime.h>
#include <math.h>

// ---------------- problem constants ----------------
#define BATCH   2048
#define KPAD    1024      // padded inner dim for both big GEMMs (actual K = 1000)
#define N1      1000
#define N2      1000
#define N3      301
#define N2PAD   1024
#define N3PAD   320       // 5 * 64
#define NSPEC   300
#define NOSC    100
#define BN_EPS  1e-5f

#define LDS_STRIDE 40     // 32 cols padded to 40 bf16 (80B) -> conflict-free b128 reads

typedef __bf16 bf16;
typedef __attribute__((ext_vector_type(16))) __bf16 v16bf;
typedef __attribute__((ext_vector_type(8)))  float  v8f;
typedef int v4i __attribute__((vector_size(16)));   // matches builtin param type

#if __has_builtin(__builtin_amdgcn_global_load_async_to_lds_b128)
#define ASYNC_COPY 1
#else
#define ASYNC_COPY 0
#endif

__device__ __forceinline__ void split_bf16(float x, bf16& h, bf16& l) {
  h = (bf16)x;
  l = (bf16)(x - (float)h);
}

// 16B global -> LDS copy; async (ASYNCcnt) when available, sync fallback otherwise.
__device__ __forceinline__ void copy16_g2l(const bf16* g, bf16* l) {
#if ASYNC_COPY
  __builtin_amdgcn_global_load_async_to_lds_b128(
      (__attribute__((address_space(1))) v4i*)(uintptr_t)g,
      (__attribute__((address_space(3))) v4i*)(uintptr_t)l,
      0, 0);
#else
  *(float4*)l = *(const float4*)g;
#endif
}

__device__ __forceinline__ void wait_async_8() {
#if ASYNC_COPY
#if __has_builtin(__builtin_amdgcn_s_wait_asynccnt)
  __builtin_amdgcn_s_wait_asynccnt(8);
#else
  asm volatile("s_wait_asynccnt 8" ::: "memory");
#endif
#endif
}
__device__ __forceinline__ void wait_async_0() {
#if ASYNC_COPY
#if __has_builtin(__builtin_amdgcn_s_wait_asynccnt)
  __builtin_amdgcn_s_wait_asynccnt(0);
#else
  asm volatile("s_wait_asynccnt 0" ::: "memory");
#endif
#endif
}

// ---------------- weight split + pad: W[fo,fi] f32 -> Whi/Wlo[fopad,KPAD] bf16 ----
__global__ void prep_weights(const float* __restrict__ W,
                             bf16* __restrict__ Whi, bf16* __restrict__ Wlo,
                             int fo, int fi, int total) {
  int idx = blockIdx.x * blockDim.x + threadIdx.x;
  if (idx >= total) return;
  int n = idx >> 10;            // / KPAD
  int k = idx & (KPAD - 1);
  float x = (n < fo && k < fi) ? W[n * fi + k] : 0.f;
  bf16 h, l; split_bf16(x, h, l);
  Whi[idx] = h; Wlo[idx] = l;
}

// ---------------- layer 1: (2048x8)@(1000x8)^T + BN + ReLU -> hi/lo bf16 padded --
__global__ void layer1_kernel(const float* __restrict__ G, const float* __restrict__ W1,
                              const float* __restrict__ b1, const float* __restrict__ gamma,
                              const float* __restrict__ beta, const float* __restrict__ mean,
                              const float* __restrict__ var,
                              bf16* __restrict__ Hhi, bf16* __restrict__ Hlo) {
  int idx = blockIdx.x * blockDim.x + threadIdx.x;  // BATCH * KPAD
  int b = idx >> 10;
  int o = idx & (KPAD - 1);
  if (b >= BATCH) return;
  float y = 0.f;
  if (o < N1) {
    const float* g = G + b * 8;
    const float* w = W1 + o * 8;
    float acc = 0.f;
#pragma unroll
    for (int i = 0; i < 8; ++i) acc = fmaf(g[i], w[i], acc);
    acc += b1[o];
    float s = gamma[o] * rsqrtf(var[o] + BN_EPS);
    y = (acc - mean[o]) * s + beta[o];
    y = fmaxf(y, 0.f);
  }
  bf16 h, l; split_bf16(y, h, l);
  Hhi[idx] = h; Hlo[idx] = l;
}

// ---------------- WMMA fragment loads from LDS tiles ----------------------------
// A 16x32 bf16 fragment: lane L(0-15): M=L, K = {c..c+7, c+16..c+23}
//                        lane L+16  : M=L, K = {c+8..c+15, c+24..c+31}
__device__ __forceinline__ v16bf ld_fragA(const bf16* p) {
  v16bf f;
  ((float4*)&f)[0] = *(const float4*)p;          // 8 bf16
  ((float4*)&f)[1] = *(const float4*)(p + 16);   // 8 bf16, +32B
  return f;
}
// B 32x16 fragment: lane L(0-15): N=L, K = {c..c+7, c+16..c+23} ; lane L+16: N=L, K = {c+8..c+15, c+24..c+31}
__device__ __forceinline__ v16bf ld_fragB(const bf16* p) {
  v16bf f;
  ((float4*)&f)[0] = *(const float4*)p;
  ((float4*)&f)[1] = *(const float4*)(p + 16);
  return f;
}

// ---------------- bf16x3 emulated-fp32 GEMM, LDS-staged, async double-buffered --
// C[M x Npad] = A[M x KPAD] @ B[Npad x KPAD]^T, block tile 64x64, 4 waves.
// mode 0: BN + ReLU  -> hi/lo bf16 at stride KPAD (zeros in pad columns)
// mode 1: BN + sigmoid -> f32 at stride outStride (masked to col < N)
__global__ void __launch_bounds__(128)
gemm_bf16x3(const bf16* __restrict__ Ahi, const bf16* __restrict__ Alo,
            const bf16* __restrict__ Bhi, const bf16* __restrict__ Blo,
            const float* __restrict__ bias, const float* __restrict__ gamma,
            const float* __restrict__ beta, const float* __restrict__ mean,
            const float* __restrict__ var,
            bf16* __restrict__ OutHi, bf16* __restrict__ OutLo,
            float* __restrict__ OutF, int outStride,
            int N, int mode) {
  __shared__ bf16 sA[2][2][64][LDS_STRIDE];   // [buf][hi/lo][row][k]
  __shared__ bf16 sB[2][2][64][LDS_STRIDE];

  const int tid   = threadIdx.x;
  const int lane  = tid & 31;
  const int wave  = tid >> 5;
  const int m0blk = blockIdx.y * 64;
  const int n0blk = blockIdx.x * 64;

  // coalesced copy slots: thread -> (row, 16-elem chunk); 2x b128 per tile per thread
  const int cr = tid >> 1;           // 0..63
  const int cc = (tid & 1) << 4;     // 0 or 16

  auto stage = [&](int buf, int k) {
    const bf16* gAh = Ahi + (size_t)(m0blk + cr) * KPAD + k + cc;
    const bf16* gAl = Alo + (size_t)(m0blk + cr) * KPAD + k + cc;
    const bf16* gBh = Bhi + (size_t)(n0blk + cr) * KPAD + k + cc;
    const bf16* gBl = Blo + (size_t)(n0blk + cr) * KPAD + k + cc;
    copy16_g2l(gAh,     &sA[buf][0][cr][cc]);
    copy16_g2l(gAh + 8, &sA[buf][0][cr][cc + 8]);
    copy16_g2l(gAl,     &sA[buf][1][cr][cc]);
    copy16_g2l(gAl + 8, &sA[buf][1][cr][cc + 8]);
    copy16_g2l(gBh,     &sB[buf][0][cr][cc]);
    copy16_g2l(gBh + 8, &sB[buf][0][cr][cc + 8]);
    copy16_g2l(gBl,     &sB[buf][1][cr][cc]);
    copy16_g2l(gBl + 8, &sB[buf][1][cr][cc + 8]);
  };

  stage(0, 0);                       // prime buffer 0

  v8f acc[4] = {};
  const int ar = wave * 16 + (lane & 15);
  const int ac = (lane >> 4) << 3;   // 0 or 8
  const int bc = (lane >> 4) << 3;   // 0 or 8

  for (int ks = 0; ks < KPAD / 32; ++ks) {
    const int buf = ks & 1;
    if (ks + 1 < KPAD / 32) {
      stage(buf ^ 1, (ks + 1) * 32); // 8 async ops into other buffer
      wait_async_8();                // previous 8 (this buffer) complete
    } else {
      wait_async_0();
    }
    __syncthreads();                 // tiles visible to all waves

    v16bf ah = ld_fragA(&sA[buf][0][ar][ac]);
    v16bf al = ld_fragA(&sA[buf][1][ar][ac]);
#pragma unroll
    for (int j = 0; j < 4; ++j) {
      const int br = j * 16 + (lane & 15);
      v16bf bh = ld_fragB(&sB[buf][0][br][bc]);
      v16bf bl = ld_fragB(&sB[buf][1][br][bc]);
      // bf16x3: A*B ~= Al*Bh + Ah*Bl + Ah*Bh   (fp32 accumulate)
      acc[j] = __builtin_amdgcn_wmma_f32_16x16x32_bf16(false, al, false, bh,
                                                       (short)0, acc[j], false, false);
      acc[j] = __builtin_amdgcn_wmma_f32_16x16x32_bf16(false, ah, false, bl,
                                                       (short)0, acc[j], false, false);
      acc[j] = __builtin_amdgcn_wmma_f32_16x16x32_bf16(false, ah, false, bh,
                                                       (short)0, acc[j], false, false);
    }
    __syncthreads();                 // done reading before buffer is overwritten
  }

  // C/D layout: vgpr i: lanes 0-15 -> (M = m0+i,   N = n0j+lane)
  //                     lanes16-31 -> (M = m0+8+i, N = n0j+lane-16)
  const int m0 = m0blk + wave * 16;
#pragma unroll
  for (int j = 0; j < 4; ++j) {
    int col = n0blk + j * 16 + (lane & 15);
    bool valid = col < N;
    float bs = 0.f, s = 0.f, mn = 0.f, bt = 0.f;
    if (valid) {
      bs = bias[col];
      s  = gamma[col] * rsqrtf(var[col] + BN_EPS);
      mn = mean[col];
      bt = beta[col];
    }
#pragma unroll
    for (int i = 0; i < 8; ++i) {
      int row = m0 + ((lane >> 4) << 3) + i;
      float v = acc[j][i];
      float y = valid ? ((v + bs - mn) * s + bt) : 0.f;
      if (mode == 0) {
        y = fmaxf(y, 0.f);
        bf16 h, l; split_bf16(y, h, l);
        size_t o = (size_t)row * KPAD + col;
        OutHi[o] = h; OutLo[o] = l;   // pad columns get zeros -> clean GEMM3 input
      } else if (valid) {
        float t = 1.f / (1.f + expf(-y));
        OutF[(size_t)row * outStride + col] = t;
      }
    }
  }
}

// ---------------- Lorentz oscillator spectral tail ------------------------------
__global__ void __launch_bounds__(128)
lorentz_kernel(const float* __restrict__ OUT, const float* __restrict__ G,
               const float* __restrict__ wgrid, float* __restrict__ T) {
  __shared__ float s_w02[NOSC], s_wp2[NOSC], s_g[NOSC], s_g2[NOSC];
  const int b = blockIdx.x;
  const int tid = threadIdx.x;

  for (int o = tid; o < NOSC; o += blockDim.x) {
    float w0 = OUT[b * N3 + 3 * o + 0] * 5.0f;
    float wp = OUT[b * N3 + 3 * o + 1] * 5.0f;
    float g  = OUT[b * N3 + 3 * o + 2] * 0.5f;
    s_w02[o] = w0 * w0;
    s_wp2[o] = wp * wp;
    s_g[o]   = g;
    s_g2[o]  = g * g;
  }
  __syncthreads();

  float d = fmaxf(fmaxf(G[b * 8 + 4], G[b * 8 + 5]),
                  fmaxf(G[b * 8 + 6], G[b * 8 + 7]));

  for (int j = tid; j < NSPEC; j += blockDim.x) {
    float wg = wgrid[j];
    float w2 = wg * wg;
    float e1 = 10.f, e2s = 0.f;
#pragma unroll 4
    for (int o = 0; o < NOSC; ++o) {
      float s1    = s_w02[o] - w2;
      float denom = fmaf(s1, s1, w2 * s_g2[o]);
      float r     = __builtin_amdgcn_rcpf(denom);
      e1  = fmaf(s_wp2[o] * s1,     r, e1);
      e2s = fmaf(s_wp2[o] * s_g[o], r, e2s);
    }
    float e2  = wg * e2s;
    float mag = sqrtf(fmaf(e1, e1, e2 * e2));
    float nn  = sqrtf(0.5f * (mag + e1));
    float kk  = sqrtf(0.5f * (mag - e1));
    float ab  = expf(-0.0005f * 4.0f * 3.14159265358979f * d * kk);
    float Tv  = 4.f * nn / fmaf(nn + 1.f, nn + 1.f, kk * kk) * ab;
    T[b * NSPEC + j] = Tv;
  }
}

// ---------------- launch ---------------------------------------------------------
extern "C" void kernel_launch(void* const* d_in, const int* in_sizes, int n_in,
                              void* d_out, int out_size, void* d_ws, size_t ws_size,
                              hipStream_t stream) {
  const float* G      = (const float*)d_in[0];
  const float* W1     = (const float*)d_in[1];
  const float* b1     = (const float*)d_in[2];
  const float* gamma1 = (const float*)d_in[3];
  const float* beta1  = (const float*)d_in[4];
  const float* mean1  = (const float*)d_in[5];
  const float* var1   = (const float*)d_in[6];
  const float* W2     = (const float*)d_in[7];
  const float* b2     = (const float*)d_in[8];
  const float* gamma2 = (const float*)d_in[9];
  const float* beta2  = (const float*)d_in[10];
  const float* mean2  = (const float*)d_in[11];
  const float* var2   = (const float*)d_in[12];
  const float* W3     = (const float*)d_in[13];
  const float* b3     = (const float*)d_in[14];
  const float* gamma3 = (const float*)d_in[15];
  const float* beta3  = (const float*)d_in[16];
  const float* mean3  = (const float*)d_in[17];
  const float* var3   = (const float*)d_in[18];
  const float* wgrid  = (const float*)d_in[19];
  float* T = (float*)d_out;

  char* ws = (char*)d_ws;
  size_t off = 0;
  auto take = [&](size_t bytes) -> char* {
    char* p = ws + off;
    off += (bytes + 255) & ~(size_t)255;
    return p;
  };
  bf16*  H1hi = (bf16*)take((size_t)BATCH * KPAD * sizeof(bf16));
  bf16*  H1lo = (bf16*)take((size_t)BATCH * KPAD * sizeof(bf16));
  bf16*  W2hi = (bf16*)take((size_t)N2PAD * KPAD * sizeof(bf16));
  bf16*  W2lo = (bf16*)take((size_t)N2PAD * KPAD * sizeof(bf16));
  bf16*  W3hi = (bf16*)take((size_t)N3PAD * KPAD * sizeof(bf16));
  bf16*  W3lo = (bf16*)take((size_t)N3PAD * KPAD * sizeof(bf16));
  bf16*  H2hi = (bf16*)take((size_t)BATCH * KPAD * sizeof(bf16));
  bf16*  H2lo = (bf16*)take((size_t)BATCH * KPAD * sizeof(bf16));
  float* OUT3 = (float*)take((size_t)BATCH * N3 * sizeof(float));
  (void)ws_size; (void)in_sizes; (void)n_in; (void)out_size;

  // 1) split/pad weights into bf16 hi/lo
  {
    int total2 = N2PAD * KPAD;
    prep_weights<<<(total2 + 255) / 256, 256, 0, stream>>>(W2, W2hi, W2lo, N2, N1, total2);
    int total3 = N3PAD * KPAD;
    prep_weights<<<(total3 + 255) / 256, 256, 0, stream>>>(W3, W3hi, W3lo, N3, N2, total3);
  }
  // 2) layer 1 (K=8) + BN + ReLU -> hi/lo bf16 (K-padded with zeros)
  {
    int total = BATCH * KPAD;
    layer1_kernel<<<(total + 255) / 256, 256, 0, stream>>>(
        G, W1, b1, gamma1, beta1, mean1, var1, H1hi, H1lo);
  }
  // 3) layer 2: 2048x1000x1000 bf16x3 WMMA GEMM + BN + ReLU
  {
    dim3 grid(N2PAD / 64, BATCH / 64);
    gemm_bf16x3<<<grid, 128, 0, stream>>>(
        H1hi, H1lo, W2hi, W2lo, b2, gamma2, beta2, mean2, var2,
        H2hi, H2lo, nullptr, 0, N2, 0);
  }
  // 4) layer 3: 2048x301x1000 bf16x3 WMMA GEMM + BN + sigmoid
  {
    dim3 grid(N3PAD / 64, BATCH / 64);
    gemm_bf16x3<<<grid, 128, 0, stream>>>(
        H2hi, H2lo, W3hi, W3lo, b3, gamma3, beta3, mean3, var3,
        nullptr, nullptr, OUT3, N3, N3, 1);
  }
  // 5) Lorentz spectral tail
  lorentz_kernel<<<BATCH, 128, 0, stream>>>(OUT3, G, wgrid, T);
}
